// TopKSAE_78185584657014
// MI455X (gfx1250) — matrix operand order repaired and checked by reading, not compiled, not run on hardware
//
#include <hip/hip_runtime.h>
#include <math.h>

// ---------------------------------------------------------------------------
// TopK SAE forward for MI455X (gfx1250, wave32, WMMA).
//   B=8192 rows, D=512 features, L=8192 latents, K=32 top-k.
// Outputs (concat in d_out, f32): x_hat [B*D], z [B*L], aux [1].
//
// Fast path: pre-split x and enc_w into bf16 hi + residual-lo planes in d_ws,
// then a WMMA GEMM whose inner loop is pure b128 loads + v_wmma (3 bf16
// products ~= fp32 accuracy). Fallback (small d_ws): fused-conversion GEMM.
// ---------------------------------------------------------------------------

typedef __attribute__((ext_vector_type(16))) __bf16 v16bf;
typedef __attribute__((ext_vector_type(8)))  __bf16 v8bf;
typedef __attribute__((ext_vector_type(8)))  float  v8f;

constexpr int B_ = 8192;
constexpr int D_ = 512;
constexpr int L_ = 8192;
constexpr int K_ = 32;

#define WMMA_BF16(A, Bm, C) \
    __builtin_amdgcn_wmma_f32_16x16x32_bf16(false, (A), false, (Bm), (short)0, (C), false, false)

// ---------------------------------------------------------------------------
// Operand loaders from pre-split row-major bf16 planes (16B aligned loads).
// A operand (16x32): per-lane 16 consecutive K values at p.
// B operand (32x16): per-lane 16 consecutive K values at p.
// ---------------------------------------------------------------------------
__device__ __forceinline__ v16bf ldA(const __bf16* __restrict__ p) {
    v8bf a = *(const v8bf*)p;
    v8bf b = *(const v8bf*)(p + 8);
    return __builtin_shufflevector(a, b, 0, 1, 2, 3, 4, 5, 6, 7,
                                   8, 9, 10, 11, 12, 13, 14, 15);
}
__device__ __forceinline__ v16bf ldB(const __bf16* __restrict__ p) {
    v8bf a = *(const v8bf*)p;
    v8bf b = *(const v8bf*)(p + 8);
    return __builtin_shufflevector(a, b, 0, 1, 2, 3, 4, 5, 6, 7,
                                   8, 9, 10, 11, 12, 13, 14, 15);
}

// ---------------------------------------------------------------------------
// Kernel P: split f32 array into bf16 hi + bf16 residual lo planes.
// One thread handles 8 elements (32B read, 2x16B writes).
// ---------------------------------------------------------------------------
__global__ __launch_bounds__(256) void split_bf16(
    const float* __restrict__ src, __bf16* __restrict__ hi,
    __bf16* __restrict__ lo, int n8) {
    int i = blockIdx.x * 256 + threadIdx.x;
    if (i >= n8) return;
    const float* p = src + (size_t)i * 8;
    float4 u = *(const float4*)p;
    float4 w = *(const float4*)(p + 4);
    float vs[8] = {u.x, u.y, u.z, u.w, w.x, w.y, w.z, w.w};
    v8bf hv, lv;
#pragma unroll
    for (int j = 0; j < 8; ++j) {
        __bf16 h = (__bf16)vs[j];
        hv[j] = h;
        lv[j] = (__bf16)(vs[j] - (float)h);
    }
    *(v8bf*)(hi + (size_t)i * 8) = hv;
    *(v8bf*)(lo + (size_t)i * 8) = lv;
}

// ---------------------------------------------------------------------------
// Kernel G (fast): pre_acts = x @ enc_w^T + enc_b from pre-split planes.
// One wave -> 32(M) x 64(N) f32 tile: 2 M-tiles x 4 N-tiles, 24 WMMA and
// 24 b128 loads per K=32 step. Writes into z region of d_out.
// ---------------------------------------------------------------------------
__global__ __launch_bounds__(256) void sae_encode_wmma_pre(
    const __bf16* __restrict__ xh, const __bf16* __restrict__ xl,
    const __bf16* __restrict__ wh, const __bf16* __restrict__ wl,
    const float* __restrict__ enc_b, float* __restrict__ z,
    float* __restrict__ aux) {
    if (blockIdx.x == 0 && threadIdx.x == 0) *aux = 0.0f;

    const int wid  = blockIdx.x * 8 + (threadIdx.x >> 5);
    const int lane = threadIdx.x & 31;
    const int r    = lane & 15;
    const int half = lane >> 4;

    const int m0 = (wid >> 7) << 5;   // 256 M-groups of 32 rows
    const int n0 = (wid & 127) << 6;  // 128 N-groups of 64 latents

    const __bf16* x0h = xh + (size_t)(m0 + r) * D_;
    const __bf16* x0l = xl + (size_t)(m0 + r) * D_;
    const __bf16* x1h = xh + (size_t)(m0 + 16 + r) * D_;
    const __bf16* x1l = xl + (size_t)(m0 + 16 + r) * D_;
    const __bf16* wrh = wh + (size_t)(n0 + r) * D_;
    const __bf16* wrl = wl + (size_t)(n0 + r) * D_;

    v8f c[2][4] = {};

    for (int kk = 0; kk < D_; kk += 32) {
        const int ao = kk + half * 16;  // A chunk base
        const int bo = kk + half * 16;  // B chunk base
        v16bf a0h = ldA(x0h + ao);
        v16bf a0l = ldA(x0l + ao);
        v16bf a1h = ldA(x1h + ao);
        v16bf a1l = ldA(x1l + ao);
#pragma unroll
        for (int j = 0; j < 4; ++j) {
            const size_t woff = (size_t)(j * 16) * D_ + bo;
            v16bf bh = ldB(wrh + woff);
            v16bf bl = ldB(wrl + woff);
            c[0][j] = WMMA_BF16(a0h, bh, c[0][j]);
            c[0][j] = WMMA_BF16(a0h, bl, c[0][j]);
            c[0][j] = WMMA_BF16(a0l, bh, c[0][j]);
            c[1][j] = WMMA_BF16(a1h, bh, c[1][j]);
            c[1][j] = WMMA_BF16(a1h, bl, c[1][j]);
            c[1][j] = WMMA_BF16(a1l, bh, c[1][j]);
        }
    }

    // C/D layout: lane holds column n = r; VGPR rr holds row m = half*8 + rr.
#pragma unroll
    for (int t = 0; t < 2; ++t) {
#pragma unroll
        for (int j = 0; j < 4; ++j) {
            const int   ncol = n0 + j * 16 + r;
            const float bias = enc_b[ncol];
            float* zp = z + (size_t)(m0 + t * 16 + half * 8) * L_ + ncol;
#pragma unroll
            for (int rr = 0; rr < 8; ++rr)
                zp[(size_t)rr * L_] = c[t][j][rr] + bias;
        }
    }
}

// ---------------------------------------------------------------------------
// Fallback fused-conversion GEMM (used when d_ws is too small).
// ---------------------------------------------------------------------------
__device__ __forceinline__ void split8(const float* __restrict__ p,
                                       v16bf& hi, v16bf& lo, int base) {
    float4 u = *(const float4*)(p);
    float4 w = *(const float4*)(p + 4);
    float vs[8] = {u.x, u.y, u.z, u.w, w.x, w.y, w.z, w.w};
#pragma unroll
    for (int i = 0; i < 8; ++i) {
        __bf16 h = (__bf16)vs[i];
        hi[base + i] = h;
        lo[base + i] = (__bf16)(vs[i] - (float)h);
    }
}

__global__ __launch_bounds__(256) void sae_encode_wmma_fused(
    const float* __restrict__ x, const float* __restrict__ enc_w,
    const float* __restrict__ enc_b, float* __restrict__ z,
    float* __restrict__ aux) {
    if (blockIdx.x == 0 && threadIdx.x == 0) *aux = 0.0f;
    const int wid  = blockIdx.x * 8 + (threadIdx.x >> 5);
    const int lane = threadIdx.x & 31;
    const int r    = lane & 15;
    const int half = lane >> 4;
    const int m0 = (wid >> 7) << 4;
    const int n0 = (wid & 127) << 6;

    const float* xrow  = x     + (size_t)(m0 + r) * D_;
    const float* wrow0 = enc_w + (size_t)(n0 + r) * D_;

    v8f c[4] = {};
    for (int kk = 0; kk < D_; kk += 32) {
        v16bf ah, al;
        split8(xrow + kk + half * 16,     ah, al, 0);
        split8(xrow + kk + half * 16 + 8, ah, al, 8);
#pragma unroll
        for (int j = 0; j < 4; ++j) {
            v16bf bh, bl;
            const float* wr = wrow0 + (size_t)(j * 16) * D_ + kk + half * 16;
            split8(wr,     bh, bl, 0);
            split8(wr + 8, bh, bl, 8);
            c[j] = WMMA_BF16(ah, bh, c[j]);
            c[j] = WMMA_BF16(ah, bl, c[j]);
            c[j] = WMMA_BF16(al, bh, c[j]);
        }
    }
#pragma unroll
    for (int j = 0; j < 4; ++j) {
        const int   ncol = n0 + j * 16 + r;
        const float bias = enc_b[ncol];
        float* zp = z + (size_t)(m0 + half * 8) * L_ + ncol;
#pragma unroll
        for (int rr = 0; rr < 8; ++rr)
            zp[(size_t)rr * L_] = c[j][rr] + bias;
    }
}

// ---------------------------------------------------------------------------
// Kernel B: per-row top-32, sparsify z in place, sparse decode, aux loss.
// One 256-thread block (8 waves) per row; row cached in 32 KB LDS.
// ---------------------------------------------------------------------------
__global__ __launch_bounds__(256) void sae_topk_decode(
    float* __restrict__ z, float* __restrict__ xhat, float* __restrict__ aux,
    const float* __restrict__ dec_w, const float* __restrict__ dec_b) {
    __shared__ float row[L_];
    __shared__ float s_val[K_];
    __shared__ int   s_idx[K_];
    __shared__ float wv[8];
    __shared__ int   wi[8];
    __shared__ float wsum[8];
    __shared__ float s_total;

    const int tid = threadIdx.x;
    const int b   = blockIdx.x;
    float* zrow = z + (size_t)b * L_;

    float sq = 0.0f;
    for (int i = tid; i < L_ / 4; i += 256) {
        float4 v = ((const float4*)zrow)[i];
        ((float4*)row)[i] = v;
        sq += v.x * v.x + v.y * v.y + v.z * v.z + v.w * v.w;
    }
    for (int off = 16; off > 0; off >>= 1) sq += __shfl_down(sq, off, 32);
    if ((tid & 31) == 0) wsum[tid >> 5] = sq;
    __syncthreads();
    if (tid == 0) {
        float t = 0.0f;
        for (int w = 0; w < 8; ++w) t += wsum[w];
        s_total = t;
    }
    __syncthreads();

    for (int it = 0; it < K_; ++it) {
        float bv = -INFINITY;
        int   bi = 0;
        for (int i = tid; i < L_; i += 256) {
            float v = row[i];
            if (v > bv) { bv = v; bi = i; }
        }
        for (int off = 16; off > 0; off >>= 1) {
            float v2 = __shfl_down(bv, off, 32);
            int   i2 = __shfl_down(bi, off, 32);
            if (v2 > bv || (v2 == bv && i2 < bi)) { bv = v2; bi = i2; }
        }
        if ((tid & 31) == 0) { wv[tid >> 5] = bv; wi[tid >> 5] = bi; }
        __syncthreads();
        if (tid == 0) {
            float best = wv[0];
            int   besti = wi[0];
            for (int w = 1; w < 8; ++w)
                if (wv[w] > best || (wv[w] == best && wi[w] < besti)) {
                    best = wv[w]; besti = wi[w];
                }
            s_val[it] = best;
            s_idx[it] = besti;
            row[besti] = -INFINITY;
        }
        __syncthreads();
    }

    const float4 zero4 = {0.0f, 0.0f, 0.0f, 0.0f};
    for (int i = tid; i < L_ / 4; i += 256) ((float4*)zrow)[i] = zero4;
    __threadfence_block();
    __syncthreads();
    if (tid < K_) zrow[s_idx[tid]] = s_val[tid];

    for (int d = tid; d < D_; d += 256) {
        float acc = dec_b[d];
        const float* wcol = dec_w + (size_t)d * L_;
#pragma unroll 8
        for (int k = 0; k < K_; ++k) acc += s_val[k] * wcol[s_idx[k]];
        xhat[(size_t)b * D_ + d] = acc;
    }

    if (tid == 0) {
        float vsq = 0.0f;
        for (int k = 0; k < K_; ++k) vsq += s_val[k] * s_val[k];
        float part = (s_total - vsq) / ((float)B_ * (float)L_);
        atomicAdd(aux, part);
    }
}

// ---------------------------------------------------------------------------
extern "C" void kernel_launch(void* const* d_in, const int* in_sizes, int n_in,
                              void* d_out, int out_size, void* d_ws,
                              size_t ws_size, hipStream_t stream) {
    (void)in_sizes; (void)n_in; (void)out_size;
    const float* x     = (const float*)d_in[0];
    const float* enc_w = (const float*)d_in[1];
    const float* enc_b = (const float*)d_in[2];
    const float* dec_w = (const float*)d_in[3];
    const float* dec_b = (const float*)d_in[4];

    float* out  = (float*)d_out;
    float* xhat = out;                                      // [B, D]
    float* z    = out + (size_t)B_ * D_;                    // [B, L]
    float* aux  = out + (size_t)B_ * D_ + (size_t)B_ * L_;  // [1]

    const size_t nx = (size_t)B_ * D_;   // x elements
    const size_t nw = (size_t)L_ * D_;   // enc_w elements
    const size_t need = (2 * nx + 2 * nw) * sizeof(__bf16);  // ~33.6 MB

    if (ws_size >= need) {
        __bf16* xh = (__bf16*)d_ws;
        __bf16* xl = xh + nx;
        __bf16* wh = xl + nx;
        __bf16* wl = wh + nw;
        split_bf16<<<(int)(nx / 8 / 256), 256, 0, stream>>>(x, xh, xl,
                                                            (int)(nx / 8));
        split_bf16<<<(int)(nw / 8 / 256), 256, 0, stream>>>(enc_w, wh, wl,
                                                            (int)(nw / 8));
        // (B/32)*(L/64) = 32768 waves -> 4096 blocks of 8 waves.
        sae_encode_wmma_pre<<<4096, 256, 0, stream>>>(xh, xl, wh, wl, enc_b, z,
                                                      aux);
    } else {
        // (B/16)*(L/64) = 65536 waves -> 8192 blocks of 8 waves.
        sae_encode_wmma_fused<<<8192, 256, 0, stream>>>(x, enc_w, enc_b, z,
                                                        aux);
    }

    sae_topk_decode<<<B_, 256, 0, stream>>>(z, xhat, aux, dec_w, dec_b);
}
